// EntropyGuidedAttention_2001454760490
// MI455X (gfx1250) — matrix operand order repaired and checked by reading, not compiled, not run on hardware
//
#include <hip/hip_runtime.h>

// ---------------------------------------------------------------------------
// EntropyGuidedAttention on MI455X (gfx1250, wave32, WMMA).
// All matmuls run on v_wmma_f32_16x16x32_bf16; f32 for LN/softmax/output.
// ---------------------------------------------------------------------------

#define BATCH 2
#define SEQ   2048
#define HID   1024
#define NHEAD 16
#define HDIM  64
#define NTOK  (BATCH * SEQ)     // 4096 tokens
#define QKVN  (3 * HID)         // 3072

typedef __bf16 bf16_t;
typedef __attribute__((ext_vector_type(16))) __bf16 v16bf;
typedef __attribute__((ext_vector_type(8)))  __bf16 v8bf;
typedef __attribute__((ext_vector_type(8)))  float  v8f;

union V16U { v16bf v; v8bf h[2]; };

__device__ __forceinline__ v8f vzero8() {
  v8f z = {0.f, 0.f, 0.f, 0.f, 0.f, 0.f, 0.f, 0.f};
  return z;
}

// A-fragment (16x32 bf16, MxK): lane holds row m = lane%16.
// lanes 0-15: K 0..7 (vgpr0-3) and 16..23 (vgpr4-7); lanes 16-31: K 8..15, 24..31.
__device__ __forceinline__ v16bf load_a_frag(const bf16_t* base, int ld, int lane) {
  const bf16_t* p = base + (size_t)(lane & 15) * ld + ((lane >> 4) & 1) * 8;
  V16U u;
  u.h[0] = *(const v8bf*)(p);
  u.h[1] = *(const v8bf*)(p + 16);
  return u.v;
}

// B-fragment (32x16 bf16, KxN): lane holds column n = lane%16.
// lanes 0-15: K 0..7 and 16..23; lanes 16-31: K 8..15, 24..31 (matches A-fragment).
// base points at column n's K=0 element; (lane&15)*ld strides columns,
// K is contiguous in memory (ld along N, unit stride along K).
__device__ __forceinline__ v16bf load_b_frag(const bf16_t* base, int ld, int lane) {
  const bf16_t* p = base + (size_t)(lane & 15) * ld + ((lane >> 4) & 1) * 8;
  V16U u;
  u.h[0] = *(const v8bf*)(p);
  u.h[1] = *(const v8bf*)(p + 16);
  return u.v;
}

__device__ __forceinline__ v8f wmma_bf16(v16bf a, v16bf b, v8f c) {
  return __builtin_amdgcn_wmma_f32_16x16x32_bf16(false, a, false, b, (short)0, c,
                                                 false, false);
}

// ---------------------------------------------------------------------------
// K0: f32 -> bf16 conversion for weights
// ---------------------------------------------------------------------------
__global__ __launch_bounds__(256) void ega_cvt_bf16(const float* __restrict__ src,
                                                    bf16_t* __restrict__ dst, int n) {
  int i = blockIdx.x * 256 + threadIdx.x;
  if (i < n) dst[i] = (bf16_t)src[i];
}

// ---------------------------------------------------------------------------
// K1: LayerNorm + entropy gate. One block (256 thr = 8 waves) per token.
// ---------------------------------------------------------------------------
__device__ __forceinline__ float block_sum256(float v, float* red) {
#pragma unroll
  for (int m = 1; m <= 16; m <<= 1) v += __shfl_xor(v, m, 32);
  if ((threadIdx.x & 31) == 0) red[threadIdx.x >> 5] = v;
  __syncthreads();
  float tot = 0.f;
#pragma unroll
  for (int i = 0; i < 8; ++i) tot += red[i];
  __syncthreads();
  return tot;
}

__global__ __launch_bounds__(256) void ega_ln_gate(
    const float* __restrict__ x, const float* __restrict__ lng,
    const float* __restrict__ lnb, const float* __restrict__ went,
    const float* __restrict__ bent, bf16_t* __restrict__ xn,
    float* __restrict__ gate) {
  __shared__ float red[8];
  const int t = blockIdx.x;
  const int tid = threadIdx.x;
  const float* xr = x + (size_t)t * HID;

  float v[4];
  float s = 0.f;
#pragma unroll
  for (int i = 0; i < 4; ++i) { v[i] = xr[tid + 256 * i]; s += v[i]; }
  s = block_sum256(s, red);
  const float mu = s * (1.0f / HID);

  float s2 = 0.f;
#pragma unroll
  for (int i = 0; i < 4; ++i) { float d = v[i] - mu; s2 += d * d; }
  s2 = block_sum256(s2, red);
  const float rstd = rsqrtf(s2 * (1.0f / HID) + 1e-6f);

  float dot = 0.f;
#pragma unroll
  for (int i = 0; i < 4; ++i) {
    const int idx = tid + 256 * i;
    const float xv = (v[i] - mu) * rstd * lng[idx] + lnb[idx];
    xn[(size_t)t * HID + idx] = (bf16_t)xv;
    dot += xv * went[idx];
  }
  dot = block_sum256(dot, red);
  if (tid == 0) {
    const float g = 1.0f / (1.0f + __expf(-(dot + bent[0])));
    gate[t] = fminf(fmaxf(g, 0.1f), 2.0f);
  }
}

// ---------------------------------------------------------------------------
// K2: QKV GEMM. C[4096,3072] = Xn(bf16) @ Wqkv^T + b. 64x64 tile per wave.
// Scatters into per-head Q[S][HD], K[S][HD], V^T[HD][S] (all bf16).
// ---------------------------------------------------------------------------
__global__ __launch_bounds__(256) void ega_qkv_gemm(
    const bf16_t* __restrict__ A, const bf16_t* __restrict__ W,
    const float* __restrict__ bias, bf16_t* __restrict__ qo,
    bf16_t* __restrict__ ko, bf16_t* __restrict__ vto) {
  const int lane = threadIdx.x & 31;
  const int wave = blockIdx.x * 8 + (threadIdx.x >> 5);
  const int NT = QKVN / 64;  // 48
  const int mt = wave / NT, nt = wave % NT;
  const int m0 = mt * 64, n0 = nt * 64;

  v8f acc[4][4];
#pragma unroll
  for (int i = 0; i < 4; ++i)
#pragma unroll
    for (int j = 0; j < 4; ++j) acc[i][j] = vzero8();

  for (int kk = 0; kk < HID; kk += 32) {
    v16bf af[4], wf[4];
#pragma unroll
    for (int i = 0; i < 4; ++i)
      af[i] = load_a_frag(A + (size_t)(m0 + 16 * i) * HID + kk, HID, lane);
#pragma unroll
    for (int j = 0; j < 4; ++j)
      wf[j] = load_b_frag(W + (size_t)(n0 + 16 * j) * HID + kk, HID, lane);
#pragma unroll
    for (int i = 0; i < 4; ++i)
#pragma unroll
      for (int j = 0; j < 4; ++j) acc[i][j] = wmma_bf16(af[i], wf[j], acc[i][j]);
  }

  const int which = n0 / HID;            // 0=q 1=k 2=v (64-col tile never spans)
  const int head = (n0 % HID) / HDIM;    // 64-col tile never spans a head
  const int half8 = ((lane >> 4) & 1) * 8;
  const int nlane = lane & 15;

#pragma unroll
  for (int i = 0; i < 4; ++i) {
#pragma unroll
    for (int j = 0; j < 4; ++j) {
      const int ncol = n0 + 16 * j + nlane;
      const float bv = bias[ncol];
      const int d = ncol & (HDIM - 1);
#pragma unroll
      for (int r = 0; r < 8; ++r) {
        const int m = m0 + 16 * i + r + half8;
        const int b = m >> 11;           // / SEQ
        const int s = m & (SEQ - 1);
        const bf16_t val = (bf16_t)(acc[i][j][r] + bv);
        const size_t hb = (size_t)(b * NHEAD + head);
        if (which == 0)      qo[(hb * SEQ + s) * HDIM + d] = val;
        else if (which == 1) ko[(hb * SEQ + s) * HDIM + d] = val;
        else                 vto[(hb * HDIM + d) * SEQ + s] = val;
      }
    }
  }
}

// ---------------------------------------------------------------------------
// K3: flash-style causal attention with post-softmax key gate.
// One wave per 16-query tile; key chunks of 32; online max/sum.
// ---------------------------------------------------------------------------
__global__ __launch_bounds__(128) void ega_attn(
    const bf16_t* __restrict__ qg, const bf16_t* __restrict__ kg,
    const bf16_t* __restrict__ vT, const float* __restrict__ gate,
    bf16_t* __restrict__ ao) {
  __shared__ __attribute__((aligned(16))) bf16_t pbuf[4][16 * 32];
  const int lane = threadIdx.x & 31;
  const int wlocal = threadIdx.x >> 5;
  const int wave = blockIdx.x * 4 + wlocal;
  const int qb = wave & 127;   // S/16 = 128 q-blocks
  const int bh = wave >> 7;    // 0..31
  const int b = bh >> 4, h = bh & 15;
  const int q0 = qb * 16;

  const bf16_t* qhead = qg + (size_t)bh * SEQ * HDIM;
  const bf16_t* khead = kg + (size_t)bh * SEQ * HDIM;
  const bf16_t* vhead = vT + (size_t)bh * HDIM * SEQ;
  const float* gb = gate + (size_t)b * SEQ;
  bf16_t* pl = &pbuf[wlocal][0];

  // Q fragments for d = 0..31 and 32..63
  v16bf qa0 = load_a_frag(qhead + (size_t)q0 * HDIM + 0, HDIM, lane);
  v16bf qa1 = load_a_frag(qhead + (size_t)q0 * HDIM + 32, HDIM, lane);

  v8f acc[4];
#pragma unroll
  for (int t = 0; t < 4; ++t) acc[t] = vzero8();

  float mrun[8], lrun[8];
#pragma unroll
  for (int r = 0; r < 8; ++r) { mrun[r] = -1e30f; lrun[r] = 0.f; }

  const int half8 = ((lane >> 4) & 1) * 8;
  const int nlane = lane & 15;
  const float sc = 1.25f;  // (1/sqrt(64)) / 0.1

  for (int j0 = 0; j0 < q0 + 16; j0 += 32) {
    // scores S = Q @ K^T  (16 x 32), two 16x16 C tiles, k-dim = HDIM
    v8f s0 = vzero8(), s1 = vzero8();
    s0 = wmma_bf16(qa0, load_b_frag(khead + (size_t)(j0)*HDIM + 0, HDIM, lane), s0);
    s0 = wmma_bf16(qa1, load_b_frag(khead + (size_t)(j0)*HDIM + 32, HDIM, lane), s0);
    s1 = wmma_bf16(qa0, load_b_frag(khead + (size_t)(j0 + 16) * HDIM + 0, HDIM, lane), s1);
    s1 = wmma_bf16(qa1, load_b_frag(khead + (size_t)(j0 + 16) * HDIM + 32, HDIM, lane), s1);

#pragma unroll
    for (int r = 0; r < 8; ++r) {
      const int qrow = q0 + r + half8;
      float a0 = s0[r] * sc;
      float a1 = s1[r] * sc;
      if (j0 + nlane > qrow) a0 = -1e30f;       // causal mask
      if (j0 + 16 + nlane > qrow) a1 = -1e30f;
      s0[r] = a0;
      s1[r] = a1;
    }

    const float g0 = gb[j0 + nlane];
    const float g1 = gb[j0 + 16 + nlane];

    float alpha[8];
#pragma unroll
    for (int r = 0; r < 8; ++r) {
      // row max across the 16-lane half holding this row's columns
      float cm = fmaxf(s0[r], s1[r]);
#pragma unroll
      for (int m = 1; m <= 8; m <<= 1) cm = fmaxf(cm, __shfl_xor(cm, m, 32));
      const float nm = fmaxf(mrun[r], cm);
      const float a = __expf(mrun[r] - nm);
      const float p0 = __expf(s0[r] - nm);
      const float p1 = __expf(s1[r] - nm);
      float rs = p0 + p1;  // ungated denominator
#pragma unroll
      for (int m = 1; m <= 8; m <<= 1) rs += __shfl_xor(rs, m, 32);
      mrun[r] = nm;
      lrun[r] = lrun[r] * a + rs;
      alpha[r] = a;
      // gate applied post-softmax (numerator only), stage P as bf16 in LDS
      const int row = r + half8;
      pl[row * 32 + nlane] = (bf16_t)(p0 * g0);
      pl[row * 32 + 16 + nlane] = (bf16_t)(p1 * g1);
    }
    asm volatile("s_wait_dscnt 0" ::: "memory");
    const v16bf pfrag = load_a_frag(pl, 32, lane);

#pragma unroll
    for (int t = 0; t < 4; ++t) {
#pragma unroll
      for (int r = 0; r < 8; ++r) acc[t][r] *= alpha[r];
      // V^T rows are contiguous in key index -> direct B-fragment load
      const v16bf vb =
          load_b_frag(vhead + (size_t)(t * 16) * SEQ + j0, SEQ, lane);
      acc[t] = wmma_bf16(pfrag, vb, acc[t]);
    }
  }

  // normalize and write merged-head activations (bf16) for the out projection
#pragma unroll
  for (int t = 0; t < 4; ++t) {
#pragma unroll
    for (int r = 0; r < 8; ++r) {
      const int m = q0 + r + half8;
      const size_t tok = (size_t)b * SEQ + m;
      const int col = h * HDIM + t * 16 + nlane;
      ao[tok * HID + col] = (bf16_t)(acc[t][r] / lrun[r]);
    }
  }
}

// ---------------------------------------------------------------------------
// K4: output projection. Y = (AO @ Wout^T + b) * 0.1, f32 output.
// ---------------------------------------------------------------------------
__global__ __launch_bounds__(256) void ega_out_gemm(
    const bf16_t* __restrict__ A, const bf16_t* __restrict__ W,
    const float* __restrict__ bias, float* __restrict__ out) {
  const int lane = threadIdx.x & 31;
  const int wave = blockIdx.x * 8 + (threadIdx.x >> 5);
  const int NT = HID / 64;  // 16
  const int mt = wave / NT, nt = wave % NT;
  const int m0 = mt * 64, n0 = nt * 64;

  v8f acc[4][4];
#pragma unroll
  for (int i = 0; i < 4; ++i)
#pragma unroll
    for (int j = 0; j < 4; ++j) acc[i][j] = vzero8();

  for (int kk = 0; kk < HID; kk += 32) {
    v16bf af[4], wf[4];
#pragma unroll
    for (int i = 0; i < 4; ++i)
      af[i] = load_a_frag(A + (size_t)(m0 + 16 * i) * HID + kk, HID, lane);
#pragma unroll
    for (int j = 0; j < 4; ++j)
      wf[j] = load_b_frag(W + (size_t)(n0 + 16 * j) * HID + kk, HID, lane);
#pragma unroll
    for (int i = 0; i < 4; ++i)
#pragma unroll
      for (int j = 0; j < 4; ++j) acc[i][j] = wmma_bf16(af[i], wf[j], acc[i][j]);
  }

  const int half8 = ((lane >> 4) & 1) * 8;
  const int nlane = lane & 15;
#pragma unroll
  for (int i = 0; i < 4; ++i) {
#pragma unroll
    for (int j = 0; j < 4; ++j) {
      const int n = n0 + 16 * j + nlane;
      const float bv = bias[n];
#pragma unroll
      for (int r = 0; r < 8; ++r) {
        const int m = m0 + 16 * i + r + half8;
        out[(size_t)m * HID + n] = (acc[i][j][r] + bv) * 0.1f;
      }
    }
  }
}

// ---------------------------------------------------------------------------
// launcher
// ---------------------------------------------------------------------------
extern "C" void kernel_launch(void* const* d_in, const int* in_sizes, int n_in,
                              void* d_out, int out_size, void* d_ws, size_t ws_size,
                              hipStream_t stream) {
  (void)in_sizes; (void)n_in; (void)out_size; (void)ws_size;
  const float* x    = (const float*)d_in[0];
  const float* ln_g = (const float*)d_in[1];
  const float* ln_b = (const float*)d_in[2];
  const float* wqkv = (const float*)d_in[3];
  const float* bqkv = (const float*)d_in[4];
  const float* went = (const float*)d_in[5];
  const float* bent = (const float*)d_in[6];
  const float* wout = (const float*)d_in[7];
  const float* bout = (const float*)d_in[8];

  char* ws = (char*)d_ws;
  // workspace layout (bytes)
  const size_t SZ_XN   = (size_t)NTOK * HID * 2;          // 8 MB
  const size_t SZ_WQKV = (size_t)QKVN * HID * 2;          // 6 MB
  const size_t SZ_WOUT = (size_t)HID * HID * 2;           // 2 MB
  const size_t SZ_GATE = (size_t)NTOK * 4;                // 16 KB
  const size_t SZ_HEADS = (size_t)BATCH * NHEAD * SEQ * HDIM * 2;  // 8 MB each

  bf16_t* xn    = (bf16_t*)(ws);
  bf16_t* wqkvb = (bf16_t*)(ws + SZ_XN);
  bf16_t* woutb = (bf16_t*)(ws + SZ_XN + SZ_WQKV);
  float*  gate  = (float*) (ws + SZ_XN + SZ_WQKV + SZ_WOUT);
  char*   p0    = ws + SZ_XN + SZ_WQKV + SZ_WOUT + SZ_GATE;
  bf16_t* qws   = (bf16_t*)(p0);
  bf16_t* kws   = (bf16_t*)(p0 + SZ_HEADS);
  bf16_t* vtws  = (bf16_t*)(p0 + 2 * SZ_HEADS);
  bf16_t* aows  = (bf16_t*)(p0 + 3 * SZ_HEADS);

  const int n_wqkv = QKVN * HID;   // 3145728
  const int n_wout = HID * HID;    // 1048576
  ega_cvt_bf16<<<(n_wqkv + 255) / 256, 256, 0, stream>>>(wqkv, wqkvb, n_wqkv);
  ega_cvt_bf16<<<(n_wout + 255) / 256, 256, 0, stream>>>(wout, woutb, n_wout);

  ega_ln_gate<<<NTOK, 256, 0, stream>>>(x, ln_g, ln_b, went, bent, xn, gate);

  // 64x64 tile per wave, 8 waves per block
  ega_qkv_gemm<<<(64 * 48) / 8, 256, 0, stream>>>(xn, wqkvb, bqkv, qws, kws, vtws);

  // one wave per 16-query tile: 32 heads * 128 q-blocks = 4096 waves, 4/block
  ega_attn<<<(32 * 128) / 4, 128, 0, stream>>>(qws, kws, vtws, gate, aows);

  ega_out_gemm<<<(64 * 16) / 8, 256, 0, stream>>>(aows, woutb, bout, (float*)d_out);
}
